// TorchGCN_77627238908321
// MI455X (gfx1250) — matrix operand order repaired and checked by reading, not compiled, not run on hardware
//
#include <hip/hip_runtime.h>
#include <hip/hip_bf16.h>

// ---------------------------------------------------------------------------
// CDNA5 (gfx1250) GCN forward pass.
// - Activations/weights pre-converted to bf16 once per layer (global bf16).
// - GEMM: 256x64 block tile, 8 waves; each wave computes 32x64 via
//   8 x v_wmma_f32_16x16x32_bf16 per K-step (2 A frags x 4 B frags,
//   B fragments reused -> 1.5 ds_load_b128 per WMMA, below LDS bank roof).
// - Graph aggregation: float4 gathers (L2-resident) + hw fp32 global atomics.
// ---------------------------------------------------------------------------

typedef __attribute__((ext_vector_type(16))) __bf16 v16bf;
typedef __attribute__((ext_vector_type(8)))  float  v8f;

#define LDS_STRIDE 40   // bf16 elems per LDS row: 80 B (16B-aligned, bank-spread)

__device__ __forceinline__ v16bf frag_2x16B(const __bf16* p0, const __bf16* p1) {
    union { uint4 u[2]; v16bf v; } u;
    u.u[0] = *(const uint4*)p0;   // ds_load_b128
    u.u[1] = *(const uint4*)p1;   // ds_load_b128
    return u.v;
}

// C[M,Ncols] = A[M,K] @ Bt[Ncols,K]^T (+bias)(+relu). K%32==0, Ncols%64==0.
// A, Bt are bf16; Bt is the weight pre-transposed: Bt[n][k] = W[k][n].
// Block: 256 threads (8 waves), 256x64 tile; wave w -> rows w*32..+31, cols 0..63.
__global__ __launch_bounds__(256)
void gemm_bf16_wmma(const __bf16* __restrict__ A, const __bf16* __restrict__ Bt,
                    const float* __restrict__ bias,
                    float* __restrict__ Cf, __bf16* __restrict__ Cb,
                    int M, int K, int Ncols, int relu) {
    __shared__ __bf16 As[256 * LDS_STRIDE];   // [m][k]  20.0 KB
    __shared__ __bf16 Bs[64  * LDS_STRIDE];   // [n][k]   5.0 KB

    const int tid     = threadIdx.x;
    const int lane    = tid & 31;
    const int wave    = tid >> 5;
    const int laneRow = lane & 15;
    const int laneHi  = lane >> 4;

    const int blockN = blockIdx.x * 64;
    const int blockM = blockIdx.y * 256;

    v8f acc[2][4] = {};   // 32x64 per wave: [m-frag][n-frag]

    for (int k0 = 0; k0 < K; k0 += 32) {
        __syncthreads();
        // ---- stage A: 256 rows x 32 k bf16, 16B chunks (4 per thread) ----
        #pragma unroll
        for (int j = 0; j < 4; ++j) {
            int idx = tid + j * 256;                 // 1024 chunks
            int r = idx >> 2, c = (idx & 3) << 3;    // c in bf16 elems
            int gr = blockM + r;
            uint4 v = (gr < M) ? *(const uint4*)(A + (size_t)gr * K + k0 + c)
                               : make_uint4(0u, 0u, 0u, 0u);
            *(uint4*)&As[r * LDS_STRIDE + c] = v;
        }
        // ---- stage B: 64 rows x 32 k bf16, 16B chunks (1 per thread) ----
        {
            int r = tid >> 2, c = (tid & 3) << 3;
            uint4 v = *(const uint4*)(Bt + (size_t)(blockN + r) * K + k0 + c);
            *(uint4*)&Bs[r * LDS_STRIDE + c] = v;
        }
        __syncthreads();

        // prefetch next A K-tile into near caches (global_prefetch_b8)
        if (k0 + 32 < K) {
            int r = tid, c = (tid & 1) << 4;         // spread over rows
            int gr = blockM + r;
            if (gr < M) __builtin_prefetch(A + (size_t)gr * K + k0 + 32 + c, 0, 3);
        }

        // ---- fragments per ISA VGPR layouts ----
        // A 16x32: lanes 0-15 hold K {0..7,16..23}; lanes 16-31 K {8..15,24..31}
        const int kSel  = laneHi * 8;
        const int aRow0 = wave * 32 + laneRow;
        v16bf afrag0 = frag_2x16B(&As[aRow0 * LDS_STRIDE + kSel],
                                  &As[aRow0 * LDS_STRIDE + kSel + 16]);
        v16bf afrag1 = frag_2x16B(&As[(aRow0 + 16) * LDS_STRIDE + kSel],
                                  &As[(aRow0 + 16) * LDS_STRIDE + kSel + 16]);
        // B 32x16: lanes 0-15 hold K {0..7,16..23} of col n; lanes 16-31 K {8..15,24..31}
        const int kOff = laneHi * 8;
        #pragma unroll
        for (int nt = 0; nt < 4; ++nt) {
            const int bRow = nt * 16 + laneRow;
            v16bf bfrag = frag_2x16B(&Bs[bRow * LDS_STRIDE + kOff],
                                     &Bs[bRow * LDS_STRIDE + kOff + 16]);
            acc[0][nt] = __builtin_amdgcn_wmma_f32_16x16x32_bf16(
                             false, afrag0, false, bfrag, (short)0, acc[0][nt], false, false);
            acc[1][nt] = __builtin_amdgcn_wmma_f32_16x16x32_bf16(
                             false, afrag1, false, bfrag, (short)0, acc[1][nt], false, false);
        }
    }

    // ---- store D: lane L, vgpr i -> row = i + 8*laneHi, col = lane&15 ----
    #pragma unroll
    for (int mt = 0; mt < 2; ++mt) {
        const int rBase = blockM + wave * 32 + mt * 16 + laneHi * 8;
        #pragma unroll
        for (int nt = 0; nt < 4; ++nt) {
            int c = blockN + nt * 16 + laneRow;
            float bv = bias ? bias[c] : 0.f;
            #pragma unroll
            for (int i = 0; i < 8; ++i) {
                int r = rBase + i;
                if (r < M) {
                    float v = acc[mt][nt][i] + bv;
                    if (relu) v = fmaxf(v, 0.f);
                    if (Cf) Cf[(size_t)r * Ncols + c] = v;
                    if (Cb) Cb[(size_t)r * Ncols + c] = (__bf16)v;
                }
            }
        }
    }
}

// ---------------------------------------------------------------------------
// Conversion kernels
// ---------------------------------------------------------------------------
__global__ void k_cvt_bf16(const float* __restrict__ x, __bf16* __restrict__ y, int n4) {
    int i = blockIdx.x * blockDim.x + threadIdx.x;
    if (i < n4) {
        float4 v = ((const float4*)x)[i];
        union { __bf16 h[4]; uint2 u; } o;
        o.h[0] = (__bf16)v.x; o.h[1] = (__bf16)v.y;
        o.h[2] = (__bf16)v.z; o.h[3] = (__bf16)v.w;
        ((uint2*)y)[i] = o.u;
    }
}

// relu(f32) -> bf16
__global__ void k_relu_cvt(const float* __restrict__ a, __bf16* __restrict__ y, int n4) {
    int i = blockIdx.x * blockDim.x + threadIdx.x;
    if (i < n4) {
        float4 v = ((const float4*)a)[i];
        union { __bf16 h[4]; uint2 u; } o;
        o.h[0] = (__bf16)fmaxf(v.x, 0.f); o.h[1] = (__bf16)fmaxf(v.y, 0.f);
        o.h[2] = (__bf16)fmaxf(v.z, 0.f); o.h[3] = (__bf16)fmaxf(v.w, 0.f);
        ((uint2*)y)[i] = o.u;
    }
}

// Wt[n][k] = (bf16) W[k][n]   (weights are tiny; coalesced read, strided write)
__global__ void k_cvt_w_t(const float* __restrict__ W, __bf16* __restrict__ Wt,
                          int K, int Nc) {
    int i = blockIdx.x * blockDim.x + threadIdx.x;
    if (i < K * Nc) {
        int k = i / Nc, n = i - k * Nc;
        Wt[(size_t)n * K + k] = (__bf16)W[i];
    }
}

// ---------------------------------------------------------------------------
// Degree / aggregation kernels (F = 512 fixed)
// ---------------------------------------------------------------------------
__global__ void k_deg_init(float* deg, int n) {
    int i = blockIdx.x * blockDim.x + threadIdx.x;
    if (i < n) deg[i] = 1.0f;                        // self-loop weight
}

__global__ void k_deg_scatter(float* deg, const int* __restrict__ dst,
                              const float* __restrict__ ew, int E) {
    int i = blockIdx.x * blockDim.x + threadIdx.x;
    if (i < E) unsafeAtomicAdd(&deg[dst[i]], ew[i]); // global_atomic_add_f32
}

__global__ void k_dinv(float* deg, int n) {
    int i = blockIdx.x * blockDim.x + threadIdx.x;
    if (i < n) deg[i] = rsqrtf(deg[i]);              // deg >= 1 always
}

// agg[r] = dinv[r]^2 * xw[r] + bias   (self-loop term, bias folded in)
__global__ __launch_bounds__(128)
void k_gcn_self(const float* __restrict__ xw, const float* __restrict__ dinv,
                const float* __restrict__ bias, float* __restrict__ agg) {
    int r = blockIdx.x, t = threadIdx.x;
    float di = dinv[r];
    float c  = di * di;
    float4 v = ((const float4*)(xw + (size_t)r * 512))[t];
    float4 b = ((const float4*)bias)[t];
    float4 o = make_float4(c * v.x + b.x, c * v.y + b.y,
                           c * v.z + b.z, c * v.w + b.w);
    ((float4*)(agg + (size_t)r * 512))[t] = o;
}

// One block (128 threads) per edge: agg[dst] += dinv[src]*ew*dinv[dst] * xw[src]
__global__ __launch_bounds__(128)
void k_gcn_scatter(const float* __restrict__ xw, float* __restrict__ agg,
                   const int* __restrict__ src, const int* __restrict__ dst,
                   const float* __restrict__ ew, const float* __restrict__ dinv) {
    int e = blockIdx.x, t = threadIdx.x;
    int s = src[e], d = dst[e];
    float coef = dinv[s] * ew[e] * dinv[d];
    float4 v = ((const float4*)(xw + (size_t)s * 512))[t];   // L2-resident gather
    float* ad = agg + (size_t)d * 512 + t * 4;
    unsafeAtomicAdd(ad + 0, coef * v.x);
    unsafeAtomicAdd(ad + 1, coef * v.y);
    unsafeAtomicAdd(ad + 2, coef * v.z);
    unsafeAtomicAdd(ad + 3, coef * v.w);
}

// ---------------------------------------------------------------------------
extern "C" void kernel_launch(void* const* d_in, const int* in_sizes, int n_in,
                              void* d_out, int out_size, void* d_ws, size_t ws_size,
                              hipStream_t stream) {
    (void)n_in; (void)out_size; (void)ws_size;
    const float* x   = (const float*)d_in[0];
    const int*   ei  = (const int*)d_in[1];    // [2, E] int32
    const float* ew  = (const float*)d_in[2];
    const float* Wi  = (const float*)d_in[3];  const float* bi  = (const float*)d_in[4];
    const float* We1 = (const float*)d_in[5];  const float* be1 = (const float*)d_in[6];
    const float* Wg0 = (const float*)d_in[7];  const float* bg0 = (const float*)d_in[8];
    const float* Wc1 = (const float*)d_in[9];  const float* bc1 = (const float*)d_in[10];
    const float* Wc2 = (const float*)d_in[11]; const float* bc2 = (const float*)d_in[12];
    const float* Wd  = (const float*)d_in[13]; const float* bd  = (const float*)d_in[14];
    const float* Wo  = (const float*)d_in[15]; const float* bo  = (const float*)d_in[16];

    const int Nn = in_sizes[0] / 128;   // 50000 nodes
    const int E  = in_sizes[2];         // 1600000 edges
    const int* src = ei;
    const int* dst = ei + E;

    // ---- workspace layout ----
    const size_t NF = (size_t)Nn * 512;
    char* w = (char*)d_ws;
    float*  xw   = (float*)w;              w += NF * sizeof(float);   // 102.4 MB
    float*  agg  = (float*)w;              w += NF * sizeof(float);   // 102.4 MB
    __bf16* act0 = (__bf16*)w;             w += NF * sizeof(__bf16);  //  51.2 MB
    __bf16* act1 = (__bf16*)w;             w += NF * sizeof(__bf16);  //  51.2 MB
    __bf16* Wt   = (__bf16*)w;             w += (size_t)512 * 512 * sizeof(__bf16);
    float*  dinv = (float*)w;

    // ---- degrees (shared by all three GCN layers) ----
    k_deg_init<<<(Nn + 255) / 256, 256, 0, stream>>>(dinv, Nn);
    k_deg_scatter<<<(E + 255) / 256, 256, 0, stream>>>(dinv, dst, ew, E);
    k_dinv<<<(Nn + 255) / 256, 256, 0, stream>>>(dinv, Nn);

    auto cvtW = [&](const float* W, int K, int Nc) {
        k_cvt_w_t<<<(K * Nc + 255) / 256, 256, 0, stream>>>(W, Wt, K, Nc);
    };
    auto gemm = [&](const __bf16* A, const float* bias, float* Cf, __bf16* Cb,
                    int M, int K, int Ncols, int relu) {
        dim3 grid(Ncols / 64, (M + 255) / 256);
        gemm_bf16_wmma<<<grid, 256, 0, stream>>>(A, Wt, bias, Cf, Cb, M, K, Ncols, relu);
    };
    auto gcn = [&](const __bf16* H, const float* W, const float* b, __bf16* actOut) {
        cvtW(W, 512, 512);
        gemm(H, nullptr, xw, nullptr, Nn, 512, 512, 0);      // xw = H @ W (f32)
        k_gcn_self<<<Nn, 128, 0, stream>>>(xw, dinv, b, agg);
        k_gcn_scatter<<<E, 128, 0, stream>>>(xw, agg, src, dst, ew, dinv);
        k_relu_cvt<<<(int)(NF / 4 + 255) / 256, 256, 0, stream>>>(agg, actOut, (int)(NF / 4));
    };

    // input x -> bf16 (N x 128), staged in act0
    k_cvt_bf16<<<(Nn * 128 / 4 + 255) / 256, 256, 0, stream>>>(x, act0, Nn * 128 / 4);

    cvtW(Wi, 128, 512);
    gemm(act0, bi, nullptr, act1, Nn, 128, 512, 1);          // input Linear + ReLU
    cvtW(We1, 512, 512);
    gemm(act1, be1, nullptr, act0, Nn, 512, 512, 1);         // encoder Linear + ReLU

    gcn(act0, Wg0, bg0, act1);                               // encoder GCNConv + ReLU
    gcn(act1, Wc1, bc1, act0);                               // conv 1 + ReLU
    gcn(act0, Wc2, bc2, act1);                               // conv 2 + ReLU

    cvtW(Wd, 512, 512);
    gemm(act1, bd, nullptr, act0, Nn, 512, 512, 1);          // decoder Linear + ReLU
    cvtW(Wo, 512, 256);
    gemm(act0, bo, (float*)d_out, nullptr, Nn, 512, 256, 0); // output Linear
}
